// TCA_82514911691267
// MI455X (gfx1250) — matrix operand order repaired and checked by reading, not compiled, not run on hardware
//
#include <hip/hip_runtime.h>
#include <hip/hip_bf16.h>

// ---------------------------------------------------------------------------
// Problem dims (fixed by the reference)
// ---------------------------------------------------------------------------
#define BQ 64
#define LQ 512
#define DW 1024
#define DI 2048
#define DLBL 512
#define NC 80

typedef __bf16 bf16_t;
typedef __attribute__((ext_vector_type(16))) __bf16 v16bf;
typedef __attribute__((ext_vector_type(8)))  __bf16 v8bf;
typedef __attribute__((ext_vector_type(4)))  __bf16 v4bf;
typedef __attribute__((ext_vector_type(8)))  float  v8f;
typedef __attribute__((ext_vector_type(4)))  int    v4i;

// Async global->LDS staging (CDNA5), guarded so a missing builtin cannot
// break the build; falls back to plain vector copies.
// Probe-confirmed prototype: (v4i AS1* src, v4i AS3* dst, int offset, int cpol)
#if defined(__has_builtin)
#  if __has_builtin(__builtin_amdgcn_global_load_async_to_lds_b128) && \
      __has_builtin(__builtin_amdgcn_s_wait_asynccnt)
#    define USE_ASYNC_LDS 1
#  endif
#endif
#ifndef USE_ASYNC_LDS
#  define USE_ASYNC_LDS 0
#endif

// ---------------------------------------------------------------------------
// Helpers
// ---------------------------------------------------------------------------
__device__ __forceinline__ float fast_tanh(float x) {
    // tanh(x) = 1 - 2/(exp2(2x*log2e) + 1); v_exp_f32 + v_rcp_f32, saturates
    float e = __builtin_amdgcn_exp2f(x * 2.88539008177792681472f);
    return 1.0f - 2.0f * __builtin_amdgcn_rcpf(e + 1.0f);
}

// A fragment: 16x32 bf16 tile (row-major source, leading dim ld).
// Lane group 0 (lanes 0-15): rows m=lane, K elems {0..15}
// Lane group 1 (lanes 16-31): rows m=lane-16, K elems {16..31}
__device__ __forceinline__ v16bf load_a_frag(const bf16_t* __restrict__ tile,
                                             int ld, int lane) {
    const int half = lane >> 4;
    const int m    = lane & 15;
    const bf16_t* row = tile + (size_t)m * ld;
    v8bf lo = *(const v8bf*)(row + half * 16);
    v8bf hi = *(const v8bf*)(row + half * 16 + 8);
    v16bf a;
#pragma unroll
    for (int i = 0; i < 8; ++i) { a[i] = lo[i]; a[i + 8] = hi[i]; }
    return a;
}

// B fragment: 32x16 bf16 (K x N) where B[k][n] = W[n0+n][k0+k], W row-major
// with leading dim ldw (i.e. W is [N_out, K] like torch Linear weights).
// Lane: column n = lane&15, K range = (lane>>4)*16 .. +15 (contiguous 32B).
__device__ __forceinline__ v16bf load_b_frag(const bf16_t* __restrict__ W,
                                             int ldw, int n0, int k0, int lane) {
    const bf16_t* p = W + (size_t)(n0 + (lane & 15)) * ldw + k0 + (lane >> 4) * 16;
    return *(const v16bf*)p;
}

// ---------------------------------------------------------------------------
// Kernel 1: fp32 -> bf16 conversion (weights)
// ---------------------------------------------------------------------------
__global__ void k_convert_bf16(const float* __restrict__ src,
                               bf16_t* __restrict__ dst, int n) {
    int i = blockIdx.x * blockDim.x + threadIdx.x;
    if (i < n) dst[i] = (bf16_t)src[i];
}

// ---------------------------------------------------------------------------
// Kernel 2: V_h = last_w * word_x, stored bf16 (vectorized x4)
// ---------------------------------------------------------------------------
__global__ void k_vh(const float4* __restrict__ wx, const float4* __restrict__ lw,
                     bf16_t* __restrict__ vh, int n4) {
    int i = blockIdx.x * blockDim.x + threadIdx.x;
    if (i < n4) {
        float4 a = wx[i], b = lw[i];
        v4bf o;
        o[0] = (bf16_t)(a.x * b.x);
        o[1] = (bf16_t)(a.y * b.y);
        o[2] = (bf16_t)(a.z * b.z);
        o[3] = (bf16_t)(a.w * b.w);
        *(v4bf*)(vh + (size_t)i * 4) = o;
    }
}

// ---------------------------------------------------------------------------
// Kernel 3: class_O = tanh(V_h @ W_label^T), bf16 out.
// M = B*L = 32768, N = 512, K = 1024.
// Each wave: one 16x64 tile (4 accumulators, A fragment reused 4x).
// K-loop fully unrolled (straight-line code, no loop-carried phis): loads
// are register-allocated directly into their consuming WMMA operand groups
// and hoisted as far ahead as pressure allows.
// ---------------------------------------------------------------------------
__global__ __launch_bounds__(256) void k_gemm_label(
        const bf16_t* __restrict__ vh, const bf16_t* __restrict__ wl,
        bf16_t* __restrict__ classO) {
    const int wid  = (blockIdx.x * blockDim.x + threadIdx.x) >> 5;
    const int lane = threadIdx.x & 31;
    const int ngrp = DLBL / 64;                  // 8 N-groups of 64
    const int tm   = wid / ngrp;
    const int n0   = (wid % ngrp) * 64;
    const bf16_t* abase = vh + (size_t)tm * 16 * DW;

    v8f acc[4] = {{}, {}, {}, {}};
#pragma unroll
    for (int k0 = 0; k0 < DW; k0 += 32) {
        v16bf a = load_a_frag(abase + k0, DW, lane);
#pragma unroll
        for (int j = 0; j < 4; ++j) {
            v16bf b = load_b_frag(wl, DW, n0 + j * 16, k0, lane);
            acc[j] = __builtin_amdgcn_wmma_f32_16x16x32_bf16(
                false, a, false, b, (short)0, acc[j], false, false);
        }
    }

    const int mbase = tm * 16 + 8 * (lane >> 4);
#pragma unroll
    for (int j = 0; j < 4; ++j) {
        const int cn = n0 + j * 16 + (lane & 15);
#pragma unroll
        for (int r = 0; r < 8; ++r)
            classO[(size_t)(mbase + r) * DLBL + cn] = (bf16_t)fast_tanh(acc[j][r]);
    }
}

// ---------------------------------------------------------------------------
// Kernel 4: raw class scores[b,c,l] = sum_e S_h[c,e] * class_O[b,l,e]
// Per batch GEMM: M = L = 512, N = C = 80, K = 512.
// Each wave: a full 16x80 stripe (5 accumulators), fully unrolled K-loop.
// Written transposed into class_attn region of d_out (softmaxed in place).
// ---------------------------------------------------------------------------
__global__ __launch_bounds__(256) void k_gemm_scores(
        const bf16_t* __restrict__ classO, const bf16_t* __restrict__ sh,
        float* __restrict__ attn) {
    const int wid  = (blockIdx.x * blockDim.x + threadIdx.x) >> 5;
    const int lane = threadIdx.x & 31;
    const int tm   = wid % (LQ / 16);            // 32 token tiles
    const int b    = wid / (LQ / 16);

    const bf16_t* abase = classO + ((size_t)b * LQ + tm * 16) * DLBL;
    v8f acc[5] = {{}, {}, {}, {}, {}};
#pragma unroll
    for (int k0 = 0; k0 < DLBL; k0 += 32) {
        v16bf a = load_a_frag(abase + k0, DLBL, lane);
#pragma unroll
        for (int j = 0; j < 5; ++j) {
            v16bf bb = load_b_frag(sh, DLBL, j * 16, k0, lane);
            acc[j] = __builtin_amdgcn_wmma_f32_16x16x32_bf16(
                false, a, false, bb, (short)0, acc[j], false, false);
        }
    }

    const int lb = tm * 16 + 8 * (lane >> 4);
    float* out = attn + (size_t)b * NC * LQ;
#pragma unroll
    for (int j = 0; j < 5; ++j) {
        const int c = j * 16 + (lane & 15);
#pragma unroll
        for (int r = 0; r < 8; ++r)
            out[(size_t)c * LQ + lb + r] = acc[j][r];
    }
}

// ---------------------------------------------------------------------------
// Kernel 5 (fused image branch): img_scores[b,l] =
//   sum_i image_x[b,i] * tanh( sum_d V_h[b,l,d] * W_image[i,d] )
// Block = (l-tile of 16, b). V_h tile staged in LDS (async, ASYNCcnt);
// each wave covers strided 64-wide i-groups with a fully unrolled K-loop;
// image_O tiles live only in registers.
// ---------------------------------------------------------------------------
__global__ __launch_bounds__(256) void k_image_scores(
        const bf16_t* __restrict__ vh, const bf16_t* __restrict__ wi,
        const float* __restrict__ image_x, float* __restrict__ img_scores) {
    __shared__ bf16_t sVh[16 * DW];              // 32 KB
    __shared__ float  sImg[DI];                  // 8 KB
    __shared__ float  sRed[8 * 16 * 16];         // 8 KB

    const int b    = blockIdx.y;
    const int lt   = blockIdx.x;
    const int tid  = threadIdx.x;
    const int wave = tid >> 5;
    const int lane = tid & 31;

    // Stage V_h tile [16 x 1024] and image_x[b,:] into LDS
    const bf16_t* vsrc = vh + ((size_t)b * LQ + lt * 16) * DW;
#if USE_ASYNC_LDS
    {
        v4i* gg = (v4i*)const_cast<bf16_t*>(vsrc);
        __attribute__((address_space(1))) v4i* g =
            (__attribute__((address_space(1))) v4i*)gg;
        __attribute__((address_space(3))) v4i* l =
            (__attribute__((address_space(3))) v4i*)(v4i*)sVh;
        // 32 KB tile = 2048 x 16B chunks; each lane supplies its own
        // global + LDS address (async copy, tracked by ASYNCcnt)
        for (int i = tid; i < 16 * DW * 2 / 16; i += 256)
            __builtin_amdgcn_global_load_async_to_lds_b128(g + i, l + i, 0, 0);
    }
#else
    {
        const v8bf* src = (const v8bf*)vsrc;
        v8bf* dstv = (v8bf*)sVh;
        for (int i = tid; i < 16 * DW / 8; i += 256) dstv[i] = src[i];
    }
#endif
    {
        const float4* isrc = (const float4*)(image_x + (size_t)b * DI);
        float4* idst = (float4*)sImg;
        for (int i = tid; i < DI / 4; i += 256) idst[i] = isrc[i];
    }
#if USE_ASYNC_LDS
    __builtin_amdgcn_s_wait_asynccnt(0);
#endif
    __syncthreads();

    float rowsum[8];
#pragma unroll
    for (int r = 0; r < 8; ++r) rowsum[r] = 0.0f;

    for (int g = wave; g < DI / 64; g += 8) {    // 32 i-groups of 64, 4/wave
        const int i0 = g * 64;
        // stream-ahead hint for the next W_image block this wave will touch
        if (g + 8 < DI / 64)
            __builtin_prefetch(wi + (size_t)(i0 + 8 * 64) * DW, 0, 1);

        v8f acc[4] = {{}, {}, {}, {}};
#pragma unroll
        for (int k0 = 0; k0 < DW; k0 += 32) {
            v16bf a = load_a_frag(sVh + k0, DW, lane);
#pragma unroll
            for (int j = 0; j < 4; ++j) {
                v16bf bb = load_b_frag(wi, DW, i0 + j * 16, k0, lane);
                acc[j] = __builtin_amdgcn_wmma_f32_16x16x32_bf16(
                    false, a, false, bb, (short)0, acc[j], false, false);
            }
        }

#pragma unroll
        for (int j = 0; j < 4; ++j) {
            const float gsc = sImg[i0 + j * 16 + (lane & 15)];
#pragma unroll
            for (int r = 0; r < 8; ++r) rowsum[r] += fast_tanh(acc[j][r]) * gsc;
        }
    }

    const int m0 = 8 * (lane >> 4), n = lane & 15;
#pragma unroll
    for (int r = 0; r < 8; ++r) sRed[wave * 256 + (m0 + r) * 16 + n] = rowsum[r];
    __syncthreads();

    if (tid < 16) {
        float s = 0.0f;
        for (int w = 0; w < 8; ++w)
            for (int n2 = 0; n2 < 16; ++n2) s += sRed[w * 256 + tid * 16 + n2];
        img_scores[(size_t)b * LQ + lt * 16 + tid] = s;
    }
}

// ---------------------------------------------------------------------------
// Kernel 6: in-place softmax over contiguous rows of length 512 (256 threads)
// ---------------------------------------------------------------------------
__global__ __launch_bounds__(256) void k_softmax512(float* __restrict__ data) {
    __shared__ float red[256];
    float* row = data + (size_t)blockIdx.x * 512;
    const int t = threadIdx.x;
    float a = row[t], b = row[t + 256];
    red[t] = fmaxf(a, b);
    __syncthreads();
    for (int s = 128; s > 0; s >>= 1) {
        if (t < s) red[t] = fmaxf(red[t], red[t + s]);
        __syncthreads();
    }
    const float m = red[0];
    __syncthreads();
    const float l2e = 1.44269504088896340736f;
    float ea = __builtin_amdgcn_exp2f((a - m) * l2e);
    float eb = __builtin_amdgcn_exp2f((b - m) * l2e);
    red[t] = ea + eb;
    __syncthreads();
    for (int s = 128; s > 0; s >>= 1) {
        if (t < s) red[t] += red[t + s];
        __syncthreads();
    }
    const float inv = __builtin_amdgcn_rcpf(red[0]);
    row[t]       = ea * inv;
    row[t + 256] = eb * inv;
}

// ---------------------------------------------------------------------------
// Kernel 7: avg_attn[b,l] = mean_c class_attn[b,c,l]
// ---------------------------------------------------------------------------
__global__ void k_avg_attn(const float* __restrict__ attn, float* __restrict__ avg) {
    int i = blockIdx.x * blockDim.x + threadIdx.x;
    if (i >= BQ * LQ) return;
    const int b = i / LQ, l = i % LQ;
    const float* p = attn + (size_t)b * NC * LQ + l;
    float s = 0.0f;
    for (int c = 0; c < NC; ++c) s += p[(size_t)c * LQ];
    avg[i] = s * (1.0f / NC);
}

// ---------------------------------------------------------------------------
// Kernel 8: class_output[b,d] = sum_l avg[b,l]  * V_h[b,l,d]
//           image_output[b,d] = sum_l imga[b,l] * V_h[b,l,d]
// One block per batch; each thread owns 4 contiguous d.
// ---------------------------------------------------------------------------
__global__ __launch_bounds__(256) void k_outputs(
        const bf16_t* __restrict__ vh, const float* __restrict__ avg,
        const float* __restrict__ imga, float* __restrict__ class_out,
        float* __restrict__ image_out) {
    __shared__ float swa[LQ];
    __shared__ float swi[LQ];
    const int b = blockIdx.x, t = threadIdx.x;
    for (int i = t; i < LQ; i += 256) {
        swa[i] = avg[(size_t)b * LQ + i];
        swi[i] = imga[(size_t)b * LQ + i];
    }
    __syncthreads();

    float accC[4] = {0, 0, 0, 0};
    float accI[4] = {0, 0, 0, 0};
    const bf16_t* base = vh + (size_t)b * LQ * DW;
    for (int l = 0; l < LQ; ++l) {
        const float wa = swa[l], wim = swi[l];
        v4bf vv = *(const v4bf*)(base + (size_t)l * DW + t * 4);
#pragma unroll
        for (int j = 0; j < 4; ++j) {
            const float v = (float)vv[j];
            accC[j] += wa * v;
            accI[j] += wim * v;
        }
    }
#pragma unroll
    for (int j = 0; j < 4; ++j) {
        class_out[(size_t)b * DW + t * 4 + j] = accC[j];
        image_out[(size_t)b * DW + t * 4 + j] = accI[j];
    }
}

// ---------------------------------------------------------------------------
// Launch
// ---------------------------------------------------------------------------
extern "C" void kernel_launch(void* const* d_in, const int* in_sizes, int n_in,
                              void* d_out, int out_size, void* d_ws, size_t ws_size,
                              hipStream_t stream) {
    const float* image_x = (const float*)d_in[0];   // [B, DI]
    const float* word_x  = (const float*)d_in[1];   // [B, L, DW]
    const float* last_w  = (const float*)d_in[2];   // [B, L, DW]
    const float* W_label = (const float*)d_in[3];   // [DLBL, DW]
    const float* W_image = (const float*)d_in[4];   // [DI, DW]
    const float* S_h     = (const float*)d_in[5];   // [NC, DLBL]

    float* class_out = (float*)d_out;                        // [B, DW]
    float* image_out = class_out + (size_t)BQ * DW;          // [B, DW]
    float* attn      = image_out + (size_t)BQ * DW;          // [B, NC, L]

    // Workspace layout (bytes, 256-aligned sections)
    char* ws = (char*)d_ws;
    bf16_t* vh_bf   = (bf16_t*)(ws);                                  // 64 MB
    bf16_t* wl_bf   = (bf16_t*)(ws + (size_t)67108864);               // 1 MB
    bf16_t* wi_bf   = (bf16_t*)(ws + (size_t)68157440);               // 4 MB
    bf16_t* sh_bf   = (bf16_t*)(ws + (size_t)72351744);               // 80 KB
    bf16_t* classO  = (bf16_t*)(ws + (size_t)72433664);               // 32 MB
    float*  img_sc  = (float*)(ws + (size_t)105988096);               // 128 KB
    float*  avg     = (float*)(ws + (size_t)106119168);               // 128 KB

    // 1) weight conversions to bf16
    k_convert_bf16<<<(DLBL * DW + 255) / 256, 256, 0, stream>>>(W_label, wl_bf, DLBL * DW);
    k_convert_bf16<<<(DI * DW + 255) / 256, 256, 0, stream>>>(W_image, wi_bf, DI * DW);
    k_convert_bf16<<<(NC * DLBL + 255) / 256, 256, 0, stream>>>(S_h, sh_bf, NC * DLBL);

    // 2) V_h bf16
    {
        const int n4 = BQ * LQ * DW / 4;
        k_vh<<<(n4 + 255) / 256, 256, 0, stream>>>((const float4*)word_x,
                                                   (const float4*)last_w, vh_bf, n4);
    }

    // 3) class_O = tanh(V_h @ W_label^T)   [32768 x 512], 16x64 per wave
    {
        const int waves = (BQ * LQ / 16) * (DLBL / 64);   // 16384
        k_gemm_label<<<waves / 8, 256, 0, stream>>>(vh_bf, wl_bf, classO);
    }

    // 4) raw class scores into attn region, 16x80 per wave
    {
        const int waves = BQ * (LQ / 16);                 // 2048
        k_gemm_scores<<<waves / 8, 256, 0, stream>>>(classO, sh_bf, attn);
    }

    // 5) fused image branch -> img_sc[b,l]
    k_image_scores<<<dim3(LQ / 16, BQ), 256, 0, stream>>>(vh_bf, wi_bf, image_x, img_sc);

    // 6) softmax class_attn rows (in place, B*NC rows of length L)
    k_softmax512<<<BQ * NC, 256, 0, stream>>>(attn);

    // 7) mean over classes
    k_avg_attn<<<(BQ * LQ + 255) / 256, 256, 0, stream>>>(attn, avg);

    // 8) softmax image scores (B rows of length L)
    k_softmax512<<<BQ, 256, 0, stream>>>(img_sc);

    // 9) both weighted sums over V_h
    k_outputs<<<BQ, 256, 0, stream>>>(vh_bf, avg, img_sc, class_out, image_out);

    (void)in_sizes; (void)n_in; (void)out_size; (void)ws_size;
}
